// GraphTransformerNet_5918464934710
// MI455X (gfx1250) — compile-verified, matrix operand order checked by reading, not run on hardware
//
#include <hip/hip_runtime.h>
#include <hip/hip_bf16.h>
#include <stdint.h>

typedef __attribute__((ext_vector_type(16))) _Float16 v16h;
typedef __attribute__((ext_vector_type(8)))  _Float16 v8h;
typedef __attribute__((ext_vector_type(4)))  _Float16 v4h;
typedef __attribute__((ext_vector_type(8)))  float    v8f;

#define N_NODES 25000
#define N_EDGES 300000
#define BGR     25
#define NPG     1000
#define DD      128
#define NHEAD   8
#define LAYERS  4

// ---------------------------------------------------------------------------
// Embedding gather: out[r,:] = table[idx[r],:]
// ---------------------------------------------------------------------------
__global__ void gather_kernel(const float* __restrict__ tab, const int* __restrict__ idx,
                              float* __restrict__ out, int rows) {
    int tid = blockIdx.x * blockDim.x + threadIdx.x;   // rows * 32 threads, 4 floats each
    if (tid >= rows * 32) return;
    int r  = tid >> 5;
    int c4 = (tid & 31) << 2;
    *(float4*)(out + (size_t)r * DD + c4) =
        *(const float4*)(tab + (size_t)idx[r] * DD + c4);
}

// ---------------------------------------------------------------------------
// Pack a KxN f32 weight matrix into WMMA-B f16 fragment layout.
// Tile (kt,nt) of 32x16; lane holds 16 contiguous halves (one v16h).
// half t of lane l -> (k = kt*32 + (t<8?t:t+8) + 8*(l>>4), n = nt*16 + (l&15))
// ---------------------------------------------------------------------------
__global__ void pack_kernel(const float* __restrict__ W, _Float16* __restrict__ out,
                            int K, int N) {
    int lane = threadIdx.x;
    int tile = blockIdx.x;
    int nnt  = N >> 4;
    int kt   = tile / nnt;
    int nt   = tile % nnt;
    int m = lane & 15, g = lane >> 4;
    int n = nt * 16 + m;
    v16h frag;
    #pragma unroll
    for (int t = 0; t < 16; ++t) {
        int k = kt * 32 + (t < 8 ? t : t + 8) + g * 8;
        frag[t] = (_Float16)W[(size_t)k * N + n];
    }
    *(v16h*)(out + ((size_t)tile * 32 + lane) * 16) = frag;
}

// ---------------------------------------------------------------------------
// Common helpers for tiled WMMA kernels (4 waves/block, 16 rows/wave)
// ---------------------------------------------------------------------------
__device__ __forceinline__ void stage_tile_f16(const float* __restrict__ A, int r0, int M,
                                               _Float16* __restrict__ lds, int lane) {
    if (r0 + 16 <= M) {
        for (int i = lane; i < 16 * 32; i += 32) {
            int rr = i >> 5, c4 = (i & 31) << 2;
            float4 v = *(const float4*)(A + (size_t)(r0 + rr) * 128 + c4);
            v4h hv = {(_Float16)v.x, (_Float16)v.y, (_Float16)v.z, (_Float16)v.w};
            *(v4h*)(&lds[rr * 128 + c4]) = hv;
        }
    } else {
        for (int i = lane; i < 16 * 32; i += 32) {
            int rr = i >> 5, c4 = (i & 31) << 2;
            int row = r0 + rr;
            float4 v = make_float4(0.f, 0.f, 0.f, 0.f);
            if (row < M) v = *(const float4*)(A + (size_t)row * 128 + c4);
            v4h hv = {(_Float16)v.x, (_Float16)v.y, (_Float16)v.z, (_Float16)v.w};
            *(v4h*)(&lds[rr * 128 + c4]) = hv;
        }
    }
}

// A fragment = two contiguous 16B runs in LDS (K-map is piecewise contiguous)
__device__ __forceinline__ v16h load_afrag(const _Float16* __restrict__ rowbase,
                                           int kt, int g) {
    v8h lo = *(const v8h*)(rowbase + kt * 32 + g * 8);
    v8h hi = *(const v8h*)(rowbase + kt * 32 + 16 + g * 8);
    return __builtin_shufflevector(lo, hi, 0, 1, 2, 3, 4, 5, 6, 7,
                                   8, 9, 10, 11, 12, 13, 14, 15);
}

// In-register LayerNorm over full 128-wide rows spread across a wave's 8
// accumulator tiles; shfl_xor masks 1/2/4/8 reduce exactly one row's 128 values.
__device__ __forceinline__ void ln_epilogue_store(v8f c[8], int lane, int r0, int M,
                                                  const float* __restrict__ gs,
                                                  const float* __restrict__ gb,
                                                  float* __restrict__ Y) {
    int m = lane & 15, g = lane >> 4;
    #pragma unroll
    for (int r = 0; r < 8; ++r) {
        float s1 = 0.f, s2 = 0.f;
        #pragma unroll
        for (int nt = 0; nt < 8; ++nt) {
            float x = c[nt][r];
            s1 += x; s2 += x * x;
        }
        #pragma unroll
        for (int o = 8; o > 0; o >>= 1) {
            s1 += __shfl_xor(s1, o, 32);
            s2 += __shfl_xor(s2, o, 32);
        }
        float mean = s1 * (1.f / 128.f);
        float var  = s2 * (1.f / 128.f) - mean * mean;
        float inv  = rsqrtf(var + 1e-5f);
        int row = r0 + r + g * 8;
        if (row < M) {
            #pragma unroll
            for (int nt = 0; nt < 8; ++nt) {
                int n = nt * 16 + m;
                Y[(size_t)row * 128 + n] = gs[n] * ((c[nt][r] - mean) * inv) + gb[n];
            }
        }
    }
}

// ---------------------------------------------------------------------------
// Plain GEMM: C[M,128] = A[M,128] @ Wp. Safe in-place (C == A).
// ---------------------------------------------------------------------------
__global__ void gemm128_kernel(const float* __restrict__ A, const _Float16* __restrict__ Bp,
                               float* __restrict__ C, int M) {
    __shared__ _Float16 As[4][16 * 128];
    int wave = threadIdx.x >> 5, lane = threadIdx.x & 31;
    int r0 = blockIdx.x * 64 + wave * 16;
    stage_tile_f16(A, r0, M, As[wave], lane);
    __syncthreads();

    int m = lane & 15, g = lane >> 4;
    const _Float16* rowbase = &As[wave][m * 128];
    v16h afr[4];
    #pragma unroll
    for (int kt = 0; kt < 4; ++kt) afr[kt] = load_afrag(rowbase, kt, g);

    bool full = (r0 + 16 <= M);
    #pragma unroll 2
    for (int nt = 0; nt < 8; ++nt) {
        int n = nt * 16 + m;
        v8f c = {0.f, 0.f, 0.f, 0.f, 0.f, 0.f, 0.f, 0.f};
        #pragma unroll
        for (int kt = 0; kt < 4; ++kt) {
            v16h bf = *(const v16h*)(Bp + ((size_t)(kt * 8 + nt) * 32 + lane) * 16);
            c = __builtin_amdgcn_wmma_f32_16x16x32_f16(false, afr[kt], false, bf,
                                                       (short)0, c, false, false);
        }
        if (full) {
            #pragma unroll
            for (int r = 0; r < 8; ++r)
                C[(size_t)(r0 + r + g * 8) * 128 + n] = c[r];
        } else {
            #pragma unroll
            for (int r = 0; r < 8; ++r) {
                int row = r0 + r + g * 8;
                if (row < M) C[(size_t)row * 128 + n] = c[r];
            }
        }
    }
}

// ---------------------------------------------------------------------------
// Fused Q/K/V: stage h once, three GEMMs against contiguous packed weights.
// ---------------------------------------------------------------------------
__global__ void gemm128_qkv_kernel(const float* __restrict__ A,
                                   const _Float16* __restrict__ Bp3, // 3 mats, 16384 halves each
                                   float* __restrict__ Q, float* __restrict__ K,
                                   float* __restrict__ V, int M) {
    __shared__ _Float16 As[4][16 * 128];
    int wave = threadIdx.x >> 5, lane = threadIdx.x & 31;
    int r0 = blockIdx.x * 64 + wave * 16;
    stage_tile_f16(A, r0, M, As[wave], lane);
    __syncthreads();

    int m = lane & 15, g = lane >> 4;
    const _Float16* rowbase = &As[wave][m * 128];
    v16h afr[4];
    #pragma unroll
    for (int kt = 0; kt < 4; ++kt) afr[kt] = load_afrag(rowbase, kt, g);

    bool full = (r0 + 16 <= M);
    #pragma unroll
    for (int mat = 0; mat < 3; ++mat) {
        float* C = (mat == 0) ? Q : (mat == 1) ? K : V;
        const _Float16* Bp = Bp3 + (size_t)mat * 16384;
        #pragma unroll 2
        for (int nt = 0; nt < 8; ++nt) {
            int n = nt * 16 + m;
            v8f c = {0.f, 0.f, 0.f, 0.f, 0.f, 0.f, 0.f, 0.f};
            #pragma unroll
            for (int kt = 0; kt < 4; ++kt) {
                v16h bf = *(const v16h*)(Bp + ((size_t)(kt * 8 + nt) * 32 + lane) * 16);
                c = __builtin_amdgcn_wmma_f32_16x16x32_f16(false, afr[kt], false, bf,
                                                           (short)0, c, false, false);
            }
            if (full) {
                #pragma unroll
                for (int r = 0; r < 8; ++r)
                    C[(size_t)(r0 + r + g * 8) * 128 + n] = c[r];
            } else {
                #pragma unroll
                for (int r = 0; r < 8; ++r) {
                    int row = r0 + r + g * 8;
                    if (row < M) C[(size_t)row * 128 + n] = c[r];
                }
            }
        }
    }
}

// ---------------------------------------------------------------------------
// Fused O-projection: Y = LN(res + A@Wp + bias). (Y may alias res.)
// ---------------------------------------------------------------------------
__global__ void gemm128_ln_kernel(const float* __restrict__ A, const _Float16* __restrict__ Bp,
                                  const float* __restrict__ bias,
                                  const float* __restrict__ res,
                                  const float* __restrict__ gs, const float* __restrict__ gb,
                                  float* __restrict__ Y, int M) {
    __shared__ _Float16 As[4][16 * 128];
    int wave = threadIdx.x >> 5, lane = threadIdx.x & 31;
    int r0 = blockIdx.x * 64 + wave * 16;
    stage_tile_f16(A, r0, M, As[wave], lane);
    __syncthreads();

    int m = lane & 15, g = lane >> 4;
    const _Float16* rowbase = &As[wave][m * 128];
    v16h afr[4];
    #pragma unroll
    for (int kt = 0; kt < 4; ++kt) afr[kt] = load_afrag(rowbase, kt, g);

    v8f c[8];
    #pragma unroll
    for (int nt = 0; nt < 8; ++nt) {
        float bv = bias[nt * 16 + m];
        c[nt] = (v8f){bv, bv, bv, bv, bv, bv, bv, bv};
        #pragma unroll
        for (int kt = 0; kt < 4; ++kt) {
            v16h bf = *(const v16h*)(Bp + ((size_t)(kt * 8 + nt) * 32 + lane) * 16);
            c[nt] = __builtin_amdgcn_wmma_f32_16x16x32_f16(false, afr[kt], false, bf,
                                                           (short)0, c[nt], false, false);
        }
    }
    // residual (all reads precede all writes; wave owns its rows exclusively)
    bool full = (r0 + 16 <= M);
    #pragma unroll
    for (int r = 0; r < 8; ++r) {
        int row = r0 + r + g * 8;
        if (full || row < M) {
            #pragma unroll
            for (int nt = 0; nt < 8; ++nt)
                c[nt][r] += res[(size_t)row * 128 + nt * 16 + m];
        }
    }
    ln_epilogue_store(c, lane, r0, M, gs, gb, Y);
}

// ---------------------------------------------------------------------------
// Fused FFN block: X = LN(X + relu(X@W1 + b1)@W2 + b2)   (128 -> 256 -> 128)
// Hidden activations live only in LDS.
// ---------------------------------------------------------------------------
__global__ void ffn_kernel(float* __restrict__ X,
                           const _Float16* __restrict__ W1p, const float* __restrict__ b1,
                           const _Float16* __restrict__ W2p, const float* __restrict__ b2,
                           const float* __restrict__ gs, const float* __restrict__ gb,
                           int M) {
    __shared__ _Float16 xs[4][16 * 128];
    __shared__ _Float16 hs[4][16 * 256];
    int wave = threadIdx.x >> 5, lane = threadIdx.x & 31;
    int r0 = blockIdx.x * 64 + wave * 16;
    stage_tile_f16(X, r0, M, xs[wave], lane);
    __syncthreads();

    int m = lane & 15, g = lane >> 4;
    v16h a1[4];
    #pragma unroll
    for (int kt = 0; kt < 4; ++kt) a1[kt] = load_afrag(&xs[wave][m * 128], kt, g);

    // GEMM1 + ReLU -> hidden (LDS, f16)
    for (int nt = 0; nt < 16; ++nt) {
        float bv = b1[nt * 16 + m];
        v8f c = {bv, bv, bv, bv, bv, bv, bv, bv};
        #pragma unroll
        for (int kt = 0; kt < 4; ++kt) {
            v16h bf = *(const v16h*)(W1p + ((size_t)(kt * 16 + nt) * 32 + lane) * 16);
            c = __builtin_amdgcn_wmma_f32_16x16x32_f16(false, a1[kt], false, bf,
                                                       (short)0, c, false, false);
        }
        #pragma unroll
        for (int r = 0; r < 8; ++r)
            hs[wave][(r + g * 8) * 256 + nt * 16 + m] = (_Float16)fmaxf(c[r], 0.f);
    }
    __syncthreads();

    v16h a2[8];
    #pragma unroll
    for (int kt = 0; kt < 8; ++kt) a2[kt] = load_afrag(&hs[wave][m * 256], kt, g);

    // GEMM2 (all 8 output tiles held for the fused LN epilogue)
    v8f c[8];
    #pragma unroll
    for (int nt = 0; nt < 8; ++nt) {
        float bv = b2[nt * 16 + m];
        c[nt] = (v8f){bv, bv, bv, bv, bv, bv, bv, bv};
        #pragma unroll
        for (int kt = 0; kt < 8; ++kt) {
            v16h bf = *(const v16h*)(W2p + ((size_t)(kt * 8 + nt) * 32 + lane) * 16);
            c[nt] = __builtin_amdgcn_wmma_f32_16x16x32_f16(false, a2[kt], false, bf,
                                                           (short)0, c[nt], false, false);
        }
    }
    // residual + fused LN
    bool full = (r0 + 16 <= M);
    #pragma unroll
    for (int r = 0; r < 8; ++r) {
        int row = r0 + r + g * 8;
        if (full || row < M) {
            #pragma unroll
            for (int nt = 0; nt < 8; ++nt)
                c[nt][r] += X[(size_t)row * 128 + nt * 16 + m];
        }
    }
    ln_epilogue_store(c, lane, r0, M, gs, gb, X);
}

// ---------------------------------------------------------------------------
// Edge attention: one wave per edge, 4 elems/lane (head = lane>>2).
// ea holds proj_e on input, score (=e_att) on output (in place).
// ---------------------------------------------------------------------------
__global__ void edge_attn_kernel(const float* __restrict__ Q, const float* __restrict__ Km,
                                 const float* __restrict__ V, float* __restrict__ ea,
                                 const int* __restrict__ src, const int* __restrict__ dst,
                                 float* __restrict__ wV, float* __restrict__ z, int E) {
    int edge = blockIdx.x * 8 + (threadIdx.x >> 5);
    int lane = threadIdx.x & 31;
    if (edge >= E) return;
    int s = src[edge], d = dst[edge];
    float4 kq = *(const float4*)(Km + (size_t)s * DD + lane * 4);
    float4 qq = *(const float4*)(Q  + (size_t)d * DD + lane * 4);
    float4 pq = *(const float4*)(ea + (size_t)edge * DD + lane * 4);
    float s0 = kq.x * qq.x * 0.25f * pq.x;
    float s1 = kq.y * qq.y * 0.25f * pq.y;
    float s2 = kq.z * qq.z * 0.25f * pq.z;
    float s3 = kq.w * qq.w * 0.25f * pq.w;
    *(float4*)(ea + (size_t)edge * DD + lane * 4) = make_float4(s0, s1, s2, s3);
    float ssum = s0 + s1 + s2 + s3;               // per-head sum over 4 lanes
    ssum += __shfl_xor(ssum, 1, 32);
    ssum += __shfl_xor(ssum, 2, 32);
    float w = __expf(fminf(fmaxf(ssum, -5.f), 5.f));
    float4 vv = *(const float4*)(V + (size_t)s * DD + lane * 4);
    float* wvd = wV + (size_t)d * DD + lane * 4;
    atomicAdd(wvd + 0, vv.x * w);
    atomicAdd(wvd + 1, vv.y * w);
    atomicAdd(wvd + 2, vv.z * w);
    atomicAdd(wvd + 3, vv.w * w);
    if ((lane & 3) == 0) atomicAdd(z + (size_t)d * NHEAD + (lane >> 2), w);
}

// h_att = wV / (z + 1e-6), in place in wV
__global__ void attn_div_kernel(float* __restrict__ wV, const float* __restrict__ z, int n) {
    int tid = blockIdx.x * blockDim.x + threadIdx.x;
    if (tid >= n * DD) return;
    int node = tid >> 7, c = tid & 127;
    wV[tid] = wV[tid] / (z[node * NHEAD + (c >> 4)] + 1e-6f);
}

// ---------------------------------------------------------------------------
// Readout: per-graph mean + tiny MLP 128->64->32->1
// ---------------------------------------------------------------------------
__global__ void graph_mean_kernel(const float* __restrict__ h, float* __restrict__ hg) {
    int gph = blockIdx.x, dd = threadIdx.x;      // 128 threads
    float s = 0.f;
    const float* p = h + (size_t)gph * NPG * DD + dd;
    for (int i = 0; i < NPG; ++i) s += p[(size_t)i * DD];
    hg[gph * DD + dd] = s * (1.f / (float)NPG);
}

__global__ void mlp_kernel(const float* __restrict__ hg,
                           const float* __restrict__ w0, const float* __restrict__ b0,
                           const float* __restrict__ w1, const float* __restrict__ b1,
                           const float* __restrict__ w2, const float* __restrict__ b2,
                           float* __restrict__ out) {
    __shared__ float y1[64];
    __shared__ float y2[32];
    int gph = blockIdx.x, t = threadIdx.x;       // 64 threads
    const float* x = hg + (size_t)gph * DD;
    float s = b0[t];
    for (int k = 0; k < 128; ++k) s += x[k] * w0[k * 64 + t];
    y1[t] = fmaxf(s, 0.f);
    __syncthreads();
    if (t < 32) {
        float s2 = b1[t];
        for (int k = 0; k < 64; ++k) s2 += y1[k] * w1[k * 32 + t];
        y2[t] = fmaxf(s2, 0.f);
    }
    __syncthreads();
    if (t == 0) {
        float s3 = b2[0];
        for (int k = 0; k < 32; ++k) s3 += y2[k] * w2[k];
        out[gph] = s3;
    }
}

// ---------------------------------------------------------------------------
extern "C" void kernel_launch(void* const* d_in, const int* in_sizes, int n_in,
                              void* d_out, int out_size, void* d_ws, size_t ws_size,
                              hipStream_t stream) {
    (void)in_sizes; (void)n_in; (void)out_size; (void)ws_size;
    const float* emb_h  = (const float*)d_in[0];
    const float* emb_e  = (const float*)d_in[1];
    const float* w_qkve = (const float*)d_in[2];   // [4][4][128][128]
    const float* o_w    = (const float*)d_in[3];   // [4][2][128][128]
    const float* o_b    = (const float*)d_in[4];   // [4][2][128]
    const float* ln_s   = (const float*)d_in[5];   // [4][4][128]
    const float* ln_bz  = (const float*)d_in[6];
    const float* ffn_w1 = (const float*)d_in[7];   // [4][2][128][256]
    const float* ffn_b1 = (const float*)d_in[8];   // [4][2][256]
    const float* ffn_w2 = (const float*)d_in[9];   // [4][2][256][128]
    const float* ffn_b2 = (const float*)d_in[10];  // [4][2][128]
    const float* mlp_w0 = (const float*)d_in[11];
    const float* mlp_b0 = (const float*)d_in[12];
    const float* mlp_w1 = (const float*)d_in[13];
    const float* mlp_b1 = (const float*)d_in[14];
    const float* mlp_w2 = (const float*)d_in[15];
    const float* mlp_b2 = (const float*)d_in[16];
    const int* tokens = (const int*)d_in[17];
    const int* etype  = (const int*)d_in[18];
    const int* srcI   = (const int*)d_in[19];
    const int* dstI   = (const int*)d_in[20];

    char* ws = (char*)d_ws;
    size_t off = 0;
    auto alloc = [&](size_t bytes) {
        size_t o = off;
        off += (bytes + 255) & ~(size_t)255;
        return o;
    };
    float* h  = (float*)(ws + alloc((size_t)N_NODES * DD * 4));
    float* e  = (float*)(ws + alloc((size_t)N_EDGES * DD * 4));
    float* Qb = (float*)(ws + alloc((size_t)N_NODES * DD * 4));
    float* Kb = (float*)(ws + alloc((size_t)N_NODES * DD * 4));
    float* Vb = (float*)(ws + alloc((size_t)N_NODES * DD * 4));
    float* pe = (float*)(ws + alloc((size_t)N_EDGES * DD * 4));   // proj_e -> e_att
    float* wV = (float*)(ws + alloc((size_t)N_NODES * DD * 4));   // segment sums -> h_att
    float* zb = (float*)(ws + alloc((size_t)N_NODES * NHEAD * 4));
    float* hg = (float*)(ws + alloc((size_t)BGR * DD * 4));
    _Float16* Wp = (_Float16*)(ws + alloc((size_t)LAYERS * 229376 * 2));

    // embeddings
    gather_kernel<<<(N_NODES * 32 + 255) / 256, 256, 0, stream>>>(emb_h, tokens, h, N_NODES);
    gather_kernel<<<(N_EDGES * 32 + 255) / 256, 256, 0, stream>>>(emb_e, etype, e, N_EDGES);

    // pack all weights to f16 WMMA-B layout
    // per-layer packed layout (halves): qkve[4]@0, o[2]@65536, ffn1[2]@98304, ffn2[2]@163840
    for (int l = 0; l < LAYERS; ++l) {
        _Float16* base = Wp + (size_t)l * 229376;
        for (int j = 0; j < 4; ++j)
            pack_kernel<<<32, 32, 0, stream>>>(w_qkve + (size_t)(l * 4 + j) * 16384,
                                               base + (size_t)j * 16384, 128, 128);
        for (int j = 0; j < 2; ++j)
            pack_kernel<<<32, 32, 0, stream>>>(o_w + (size_t)(l * 2 + j) * 16384,
                                               base + 65536 + (size_t)j * 16384, 128, 128);
        for (int j = 0; j < 2; ++j)
            pack_kernel<<<64, 32, 0, stream>>>(ffn_w1 + (size_t)(l * 2 + j) * 32768,
                                               base + 98304 + (size_t)j * 32768, 128, 256);
        for (int j = 0; j < 2; ++j)
            pack_kernel<<<64, 32, 0, stream>>>(ffn_w2 + (size_t)(l * 2 + j) * 32768,
                                               base + 163840 + (size_t)j * 32768, 256, 128);
    }

    int gN = (N_NODES + 63) / 64;
    int gE = (N_EDGES + 63) / 64;

    for (int l = 0; l < LAYERS; ++l) {
        const _Float16* base = Wp + (size_t)l * 229376;
        // Q,K,V fused (h staged once) + proj_e
        gemm128_qkv_kernel<<<gN, 128, 0, stream>>>(h, base, Qb, Kb, Vb, N_NODES);
        gemm128_kernel<<<gE, 128, 0, stream>>>(e, base + 49152, pe, N_EDGES);
        // attention
        hipMemsetAsync(wV, 0, (size_t)N_NODES * DD * 4, stream);
        hipMemsetAsync(zb, 0, (size_t)N_NODES * NHEAD * 4, stream);
        edge_attn_kernel<<<(N_EDGES + 7) / 8, 256, 0, stream>>>(Qb, Kb, Vb, pe, srcI, dstI,
                                                                wV, zb, N_EDGES);
        attn_div_kernel<<<(N_NODES * DD + 255) / 256, 256, 0, stream>>>(wV, zb, N_NODES);
        // O projections fused with residual + LN
        gemm128_ln_kernel<<<gN, 128, 0, stream>>>(wV, base + 65536,
                                                  o_b  + (size_t)(l * 2 + 0) * 128,
                                                  h,
                                                  ln_s + (size_t)(l * 4 + 0) * 128,
                                                  ln_bz + (size_t)(l * 4 + 0) * 128,
                                                  h, N_NODES);
        gemm128_ln_kernel<<<gE, 128, 0, stream>>>(pe, base + 65536 + 16384,
                                                  o_b  + (size_t)(l * 2 + 1) * 128,
                                                  e,
                                                  ln_s + (size_t)(l * 4 + 1) * 128,
                                                  ln_bz + (size_t)(l * 4 + 1) * 128,
                                                  e, N_EDGES);
        // FFNs fused with residual + LN
        ffn_kernel<<<gN, 128, 0, stream>>>(h, base + 98304,
                                           ffn_b1 + (size_t)(l * 2 + 0) * 256,
                                           base + 163840,
                                           ffn_b2 + (size_t)(l * 2 + 0) * 128,
                                           ln_s + (size_t)(l * 4 + 2) * 128,
                                           ln_bz + (size_t)(l * 4 + 2) * 128, N_NODES);
        ffn_kernel<<<gE, 128, 0, stream>>>(e, base + 98304 + 32768,
                                           ffn_b1 + (size_t)(l * 2 + 1) * 256,
                                           base + 163840 + 32768,
                                           ffn_b2 + (size_t)(l * 2 + 1) * 128,
                                           ln_s + (size_t)(l * 4 + 3) * 128,
                                           ln_bz + (size_t)(l * 4 + 3) * 128, N_EDGES);
    }

    // readout
    graph_mean_kernel<<<BGR, 128, 0, stream>>>(h, hg);
    mlp_kernel<<<BGR, 64, 0, stream>>>(hg, mlp_w0, mlp_b0, mlp_w1, mlp_b1, mlp_w2, mlp_b2,
                                       (float*)d_out);
}
